// FrameTaggingTransformerWithCRF_50946902065810
// MI455X (gfx1250) — matrix operand order repaired and checked by reading, not compiled, not run on hardware
//
#include <hip/hip_runtime.h>
#include <math.h>

// ---------------------------------------------------------------------------
// FrameTaggingTransformerWithCRF for MI455X (gfx1250, wave32, WMMA).
// One workgroup (256 threads = 8 waves) per batch element; hidden state and
// all intermediates live in LDS; every GEMM runs on v_wmma_f32_16x16x32_f16;
// per-layer weights are streamed into LDS with global_load_async_to_lds_b128
// (ASYNCcnt path) before being converted to f16 operand buffers.
// ---------------------------------------------------------------------------

typedef _Float16 h16;
typedef __attribute__((ext_vector_type(16))) _Float16 v16h;
typedef __attribute__((ext_vector_type(8)))  float    v8f;

#define SEQ    256
#define DM     32
#define NTAGS  16

struct LayerP {
  const float *W1, *W2, *Wo, *Wqkv, *b1, *b2, *bo, *bqkv;
  const float *ln1_b, *ln1_g, *ln2_b, *ln2_g;
};
struct P {
  const float *ssim, *flags;
  const float *crf_end, *crf_start, *crf_trans, *in_W, *in_b;
  LayerP layer[2];
  const float *op_emb, *tag_W, *tag_b;
  const int *op_types, *labels;
  float *out;
};

// ---- CDNA5 async copy: global -> LDS, tracked by ASYNCcnt ------------------

__device__ __forceinline__ void async_load_b128(unsigned lds_off,
                                                const void* gptr) {
  unsigned long long ga = (unsigned long long)gptr;
  asm volatile("global_load_async_to_lds_b128 %0, %1, off"
               :: "v"(lds_off), "v"(ga) : "memory");
}
__device__ __forceinline__ void wait_async0() {
  asm volatile("s_wait_asynccnt 0" ::: "memory");
}
// Generic shared-pointer -> LDS byte offset (flat LDS aperture: low 32 bits).
__device__ __forceinline__ unsigned lds_off_of(const void* p) {
  return (unsigned)(unsigned long long)p;
}

// ---- WMMA helpers ----------------------------------------------------------

__device__ __forceinline__ v8f wmma_f16(v16h a, v16h b, v8f c) {
  return __builtin_amdgcn_wmma_f32_16x16x32_f16(false, a, false, b,
                                                (short)0, c, false, false);
}

// A-matrix (16x32 f16) lane layout: lanes0-15 row M=lane, K={0..7,16..23};
// lanes16-31 row M=lane-16, K={8..15,24..31}. 2 K-values per VGPR slot.
__device__ __forceinline__ v16h ldA_f32(const float* base, int mStride) {
  int lane = threadIdx.x & 31;
  int m = lane & 15, hh = lane >> 4;
  v16h r;
#pragma unroll
  for (int i = 0; i < 8; ++i) {
    int kb = (i < 4 ? 2 * i : 16 + 2 * (i - 4)) + hh * 8;
    r[2 * i]     = (h16)base[m * mStride + kb];
    r[2 * i + 1] = (h16)base[m * mStride + kb + 1];
  }
  return r;
}

__device__ __forceinline__ v16h ldA_h(const h16* base, int mStride, int kmax) {
  int lane = threadIdx.x & 31;
  int m = lane & 15, hh = lane >> 4;
  v16h r;
#pragma unroll
  for (int i = 0; i < 8; ++i) {
    int kb = (i < 4 ? 2 * i : 16 + 2 * (i - 4)) + hh * 8;
    r[2 * i]     = (kb     < kmax) ? base[m * mStride + kb]     : (h16)0.f;
    r[2 * i + 1] = (kb + 1 < kmax) ? base[m * mStride + kb + 1] : (h16)0.f;
  }
  return r;
}

// B-matrix (32x16 f16, KxN): lanes0-15 N=lane K={0..7,16..23}, lanes16-31 K={8..15,24..31}.
__device__ __forceinline__ v16h ldB_h(const h16* base, int kStride, int nStride,
                                      int kmax) {
  int lane = threadIdx.x & 31;
  int n = lane & 15, hh = lane >> 4;
  v16h r;
#pragma unroll
  for (int j = 0; j < 8; ++j) {
    int k = (j < 4 ? 2 * j : 16 + 2 * (j - 4)) + hh * 8;
    r[2 * j]     = (k     < kmax) ? base[k * kStride + n * nStride]       : (h16)0.f;
    r[2 * j + 1] = (k + 1 < kmax) ? base[(k + 1) * kStride + n * nStride] : (h16)0.f;
  }
  return r;
}

// 16-lane-half reductions (wave32; masks 1..8 stay inside each half).
__device__ __forceinline__ float halfmax16(float v) {
#pragma unroll
  for (int m = 1; m < 16; m <<= 1) v = fmaxf(v, __shfl_xor(v, m, 32));
  return v;
}
__device__ __forceinline__ float halfsum16(float v) {
#pragma unroll
  for (int m = 1; m < 16; m <<= 1) v += __shfl_xor(v, m, 32);
  return v;
}

// ---------------------------------------------------------------------------

__global__ void __launch_bounds__(256)
transformer_crf_kernel(P p) {
  __shared__ float xs[SEQ * DM];                         // 32 KB hidden state
  __shared__ __align__(16) unsigned char scr_raw[32768]; // wstage|qkv|ffh|emis
  __shared__ h16 oh[SEQ * DM];                           // 16 KB attn/pre-LN
  __shared__ h16 wqkv[96 * 32], wo[32 * 32], w1[64 * 32], w2[32 * 64];
  __shared__ h16 tagW[16 * 32];
  __shared__ h16 pw[8 * 16 * 16];                        // per-wave staging
  __shared__ float red[256];
  __shared__ float bqkv[96], bo_[32], b1_[64], b2_[32];
  __shared__ float ln1g[32], ln1b[32], ln2g[32], ln2b[32];
  __shared__ float tagb[16], crfs[16], crfe[16], crft[256];

  const int tid = threadIdx.x, wave = tid >> 5, lane = tid & 31;
  const int b = blockIdx.x;
  h16*   scrH = (h16*)scr_raw;
  float* scrF = (float*)scr_raw;

  // Prefetch this block's streaming inputs (lowers to global_prefetch_b8).
  __builtin_prefetch(p.labels + (size_t)b * SEQ + tid, 0, 3);
  __builtin_prefetch(p.ssim + ((size_t)b * SEQ + tid) * 6, 0, 3);

  if (tid < 16) { crfs[tid] = p.crf_start[tid]; crfe[tid] = p.crf_end[tid];
                  tagb[tid] = p.tag_b[tid]; }
  crft[tid] = p.crf_trans[tid];
  for (int i = tid; i < 16 * 32; i += 256) tagW[i] = (h16)p.tag_W[i];

  // ---- Stage 0: input projection * sqrt(dm) + positional encoding ----------
  {
    const int s = tid;
    float f[15];
    const float* sp = p.ssim + ((size_t)b * SEQ + s) * 6;
#pragma unroll
    for (int i = 0; i < 6; ++i) f[i] = sp[i];
    const int ot = p.op_types[b];
#pragma unroll
    for (int i = 0; i < 8; ++i) f[6 + i] = p.op_emb[ot * 8 + i];
    f[14] = p.flags[(size_t)b * SEQ + s];
    const float sq = 5.656854249f;               // sqrt(32)
    for (int d = 0; d < DM; ++d) {
      float acc = p.in_b[d];
      const float* w = p.in_W + d * 15;
#pragma unroll
      for (int i = 0; i < 15; ++i) acc += f[i] * w[i];
      const int i2 = d >> 1;
      const float div = expf(-(float)(2 * i2) * (9.210340372f / 32.f));
      const float ang = (float)s * div;
      const float pe = (d & 1) ? cosf(ang) : sinf(ang);
      xs[s * DM + d] = acc * sq + pe;
    }
  }
  __syncthreads();

  // ---- Encoder layers -------------------------------------------------------
  for (int L = 0; L < 2; ++L) {
    const LayerP ly = p.layer[L];

    // -- Async DMA of all f32 weights (exactly 32 KB) into the scratch LDS --
    // Layout in stage: [0,3072) Wqkv | [3072,4096) Wo | [4096,6144) W1
    //                  | [6144,8192) W2   (floats)
    {
      const unsigned base_off = lds_off_of(scrF);
      // Wqkv: 768 x 16B chunks
      for (int c = tid; c < 768; c += 256)
        async_load_b128(base_off + c * 16, ly.Wqkv + c * 4);
      // Wo: 256 chunks
      for (int c = tid; c < 256; c += 256)
        async_load_b128(base_off + 3072 * 4 + c * 16, ly.Wo + c * 4);
      // W1: 512 chunks
      for (int c = tid; c < 512; c += 256)
        async_load_b128(base_off + 4096 * 4 + c * 16, ly.W1 + c * 4);
      // W2: 512 chunks
      for (int c = tid; c < 512; c += 256)
        async_load_b128(base_off + 6144 * 4 + c * 16, ly.W2 + c * 4);
    }
    // Biases / LN params via normal loads while the DMA is in flight.
    if (tid < 96) bqkv[tid] = ly.bqkv[tid];
    if (tid < 64) b1_[tid]  = ly.b1[tid];
    if (tid < 32) { bo_[tid] = ly.bo[tid];  b2_[tid] = ly.b2[tid];
                    ln1g[tid] = ly.ln1_g[tid]; ln1b[tid] = ly.ln1_b[tid];
                    ln2g[tid] = ly.ln2_g[tid]; ln2b[tid] = ly.ln2_b[tid]; }
    wait_async0();
    __syncthreads();
    // Convert staged f32 weights to f16 operand buffers.
    for (int i = tid; i < 96 * 32; i += 256) wqkv[i] = (h16)scrF[i];
    for (int i = tid; i < 32 * 32; i += 256) wo[i]   = (h16)scrF[3072 + i];
    for (int i = tid; i < 64 * 32; i += 256) w1[i]   = (h16)scrF[4096 + i];
    for (int i = tid; i < 32 * 64; i += 256) w2[i]   = (h16)scrF[6144 + i];
    __syncthreads();

    h16* qh = scrH;               // [256][16] per head-pair (reuses stage)
    h16* kh = scrH + 4096;
    h16* vh = scrH + 8192;

    // -------- Attention, two heads (one 16-wide column pair) at a time ------
    for (int pr = 0; pr < 2; ++pr) {
      // QKV GEMM: 3 mats x 16 M-tiles, one K=32 WMMA each.
      for (int t = wave; t < 48; t += 8) {
        const int mat = t >> 4, mt = t & 15;
        v16h A = ldA_f32(xs + mt * 16 * DM, DM);
        v16h B = ldB_h(wqkv + (mat * 32 + pr * 16) * 32, 1, 32, 32);
        v8f acc = {};
        acc = wmma_f16(A, B, acc);
        h16* dst = (mat == 0 ? qh : (mat == 1 ? kh : vh));
        const int n = lane & 15, mh = (lane >> 4) * 8;
        const float bv = bqkv[mat * 32 + pr * 16 + n];
#pragma unroll
        for (int r = 0; r < 8; ++r)
          dst[(mt * 16 + mh + r) * 16 + n] = (h16)(acc[r] + bv);
      }
      __syncthreads();

      // Flash-style attention per head of this pair.
      const float scale = 0.3535533906f;          // 1/sqrt(8)
      for (int qi = 0; qi < 2; ++qi) {
        const int head = pr * 2 + qi, c0 = qi * 8;
        for (int ti = 0; ti < 2; ++ti) {
          const int mt = wave + ti * 8;
          v16h Aq = ldA_h(qh + mt * 16 * 16 + c0, 16, 8);
          float mrow[8], lrow[8];
          v8f oacc = {};
#pragma unroll
          for (int r = 0; r < 8; ++r) { mrow[r] = -3.0e38f; lrow[r] = 0.f; }
          for (int nt = 0; nt < 16; ++nt) {
            v16h Bk = ldB_h(kh + nt * 16 * 16 + c0, 1, 16, 8);
            v8f sc = {};
            sc = wmma_f16(Aq, Bk, sc);
            h16* pws = pw + wave * 256;
            const int n = lane & 15, mh = (lane >> 4) * 8;
#pragma unroll
            for (int r = 0; r < 8; ++r) {
              const float v   = sc[r] * scale;
              const float nm  = fmaxf(mrow[r], halfmax16(v));
              const float cor = expf(mrow[r] - nm);
              const float pv  = expf(v - nm);
              lrow[r] = lrow[r] * cor + halfsum16(pv);
              oacc[r] *= cor;
              mrow[r] = nm;
              pws[(mh + r) * 16 + n] = (h16)pv;
            }
            v16h Ap = ldA_h(pws, 16, 16);
            v16h Bv = ldB_h(vh + nt * 16 * 16 + c0, 16, 1, 16);
            oacc = wmma_f16(Ap, Bv, oacc);
          }
          const int n = lane & 15, mh = (lane >> 4) * 8;
          if (n < 8) {
#pragma unroll
            for (int r = 0; r < 8; ++r)
              oh[(mt * 16 + mh + r) * DM + head * 8 + n] =
                  (h16)(oacc[r] / lrow[r]);
          }
        }
      }
      __syncthreads();
    }

    // -------- Output projection + residual (rows exclusive per wave) --------
    for (int ti = 0; ti < 2; ++ti) {
      const int mt = wave + ti * 8;
      v16h A  = ldA_h(oh + mt * 16 * DM, DM, 32);
      v16h B0 = ldB_h(wo, 1, 32, 32);
      v16h B1 = ldB_h(wo + 16 * 32, 1, 32, 32);
      v8f a0 = {}, a1 = {};
      a0 = wmma_f16(A, B0, a0);
      a1 = wmma_f16(A, B1, a1);
      const int n = lane & 15, mh = (lane >> 4) * 8;
#pragma unroll
      for (int r = 0; r < 8; ++r) {
        const int row = mt * 16 + mh + r;
        oh[row * DM + n]      = (h16)(a0[r] + bo_[n]      + xs[row * DM + n]);
        oh[row * DM + 16 + n] = (h16)(a1[r] + bo_[16 + n] + xs[row * DM + 16 + n]);
      }
    }
    __syncthreads();
    { // LayerNorm 1 (one row per thread)
      const int s = tid;
      float mu = 0.f, var = 0.f, v[DM];
      for (int d = 0; d < DM; ++d) { v[d] = (float)oh[s * DM + d]; mu += v[d]; }
      mu *= (1.f / DM);
      for (int d = 0; d < DM; ++d) { float t = v[d] - mu; var += t * t; }
      var *= (1.f / DM);
      const float rs = rsqrtf(var + 1e-5f);
      for (int d = 0; d < DM; ++d)
        xs[s * DM + d] = (v[d] - mu) * rs * ln1g[d] + ln1b[d];
    }
    __syncthreads();

    // -------- FFN: x@W1.T -> GELU -> ffh -------------------------------------
    h16* ffh = scrH;              // [256][64]
    for (int t = wave; t < 64; t += 8) {
      const int mt = t >> 2, nt = t & 3;
      v16h A = ldA_f32(xs + mt * 16 * DM, DM);
      v16h B = ldB_h(w1 + nt * 16 * 32, 1, 32, 32);
      v8f acc = {};
      acc = wmma_f16(A, B, acc);
      const int n = lane & 15, mh = (lane >> 4) * 8;
#pragma unroll
      for (int r = 0; r < 8; ++r) {
        float v = acc[r] + b1_[nt * 16 + n];
        v = 0.5f * v * (1.f + erff(v * 0.7071067812f));   // exact GELU
        ffh[(mt * 16 + mh + r) * 64 + nt * 16 + n] = (h16)v;
      }
    }
    __syncthreads();

    // -------- FFN: ffh@W2.T + residual (rows exclusive per wave) ------------
    for (int ti = 0; ti < 2; ++ti) {
      const int mt = wave + ti * 8;
      v16h A0 = ldA_h(ffh + mt * 16 * 64, 64, 32);
      v16h A1 = ldA_h(ffh + mt * 16 * 64 + 32, 64, 32);
#pragma unroll
      for (int nt = 0; nt < 2; ++nt) {
        v16h B0 = ldB_h(w2 + nt * 16 * 64, 1, 64, 32);
        v16h B1 = ldB_h(w2 + nt * 16 * 64 + 32, 1, 64, 32);
        v8f acc = {};
        acc = wmma_f16(A0, B0, acc);
        acc = wmma_f16(A1, B1, acc);
        const int n = lane & 15, mh = (lane >> 4) * 8;
#pragma unroll
        for (int r = 0; r < 8; ++r) {
          const int row = mt * 16 + mh + r, col = nt * 16 + n;
          oh[row * DM + col] = (h16)(acc[r] + b2_[col] + xs[row * DM + col]);
        }
      }
    }
    __syncthreads();
    { // LayerNorm 2
      const int s = tid;
      float mu = 0.f, var = 0.f, v[DM];
      for (int d = 0; d < DM; ++d) { v[d] = (float)oh[s * DM + d]; mu += v[d]; }
      mu *= (1.f / DM);
      for (int d = 0; d < DM; ++d) { float t = v[d] - mu; var += t * t; }
      var *= (1.f / DM);
      const float rs = rsqrtf(var + 1e-5f);
      for (int d = 0; d < DM; ++d)
        xs[s * DM + d] = (v[d] - mu) * rs * ln2g[d] + ln2b[d];
    }
    __syncthreads();
  }

  // ---- Emissions: x @ tag_W.T + tag_b --------------------------------------
  float* em = scrF;               // [256][16] f32
  for (int ti = 0; ti < 2; ++ti) {
    const int mt = wave + ti * 8;
    v16h A = ldA_f32(xs + mt * 16 * DM, DM);
    v16h B = ldB_h(tagW, 1, 32, 32);
    v8f acc = {};
    acc = wmma_f16(A, B, acc);
    const int n = lane & 15, mh = (lane >> 4) * 8;
#pragma unroll
    for (int r = 0; r < 8; ++r)
      em[(mt * 16 + mh + r) * NTAGS + n] = acc[r] + tagb[n];
  }
  __syncthreads();

  // ---- CRF: gold path score (parallel) -------------------------------------
  const int* tg = p.labels + (size_t)b * SEQ;
  {
    const int s = tid;
    const int t = tg[s];
    float part = em[s * NTAGS + t];
    part += (s == 0) ? crfs[t] : crft[tg[s - 1] * NTAGS + t];
    if (s == SEQ - 1) part += crfe[t];
    red[tid] = part;
  }
  __syncthreads();
  for (int off = 128; off > 0; off >>= 1) {
    if (tid < off) red[tid] += red[tid + off];
    __syncthreads();
  }
  const float num = red[0];

  // ---- CRF: forward algorithm (wave 0, one lane per tag) -------------------
  if (wave == 0) {
    const int j = lane & 15;
    float sc = crfs[j] + em[j];
    float trc[16];
#pragma unroll
    for (int i = 0; i < 16; ++i) trc[i] = crft[i * NTAGS + j];
    for (int t = 1; t < SEQ; ++t) {
      float tv[16], m = -3.0e38f;
#pragma unroll
      for (int i = 0; i < 16; ++i) {
        tv[i] = __shfl(sc, i, 32) + trc[i];
        m = fmaxf(m, tv[i]);
      }
      float ss = 0.f;
#pragma unroll
      for (int i = 0; i < 16; ++i) ss += expf(tv[i] - m);
      sc = m + logf(ss) + em[t * NTAGS + j];
    }
    const float v  = sc + crfe[j];
    const float mm = halfmax16(v);
    const float se = halfsum16(expf(v - mm));
    const float den = mm + logf(se);
    if (lane == 0) atomicAdd(p.out, (den - num) * (1.f / 1024.f));
  }
}

__global__ void zero_out_kernel(float* o) {
  if (threadIdx.x == 0) o[0] = 0.f;
}

// ---------------------------------------------------------------------------

extern "C" void kernel_launch(void* const* d_in, const int* in_sizes, int n_in,
                              void* d_out, int out_size, void* d_ws, size_t ws_size,
                              hipStream_t stream) {
  (void)in_sizes; (void)n_in; (void)out_size; (void)d_ws; (void)ws_size;
  P p;
  p.ssim      = (const float*)d_in[0];
  p.flags     = (const float*)d_in[1];
  p.crf_end   = (const float*)d_in[2];
  p.crf_start = (const float*)d_in[3];
  p.crf_trans = (const float*)d_in[4];
  p.in_W      = (const float*)d_in[5];
  p.in_b      = (const float*)d_in[6];
  for (int l = 0; l < 2; ++l) {
    const int base = 7 + l * 12;     // sorted leaf order within each layer dict
    p.layer[l].W1    = (const float*)d_in[base + 0];
    p.layer[l].W2    = (const float*)d_in[base + 1];
    p.layer[l].Wo    = (const float*)d_in[base + 2];
    p.layer[l].Wqkv  = (const float*)d_in[base + 3];
    p.layer[l].b1    = (const float*)d_in[base + 4];
    p.layer[l].b2    = (const float*)d_in[base + 5];
    p.layer[l].bo    = (const float*)d_in[base + 6];
    p.layer[l].bqkv  = (const float*)d_in[base + 7];
    p.layer[l].ln1_b = (const float*)d_in[base + 8];
    p.layer[l].ln1_g = (const float*)d_in[base + 9];
    p.layer[l].ln2_b = (const float*)d_in[base + 10];
    p.layer[l].ln2_g = (const float*)d_in[base + 11];
  }
  p.op_emb   = (const float*)d_in[31];
  p.tag_W    = (const float*)d_in[32];
  p.tag_b    = (const float*)d_in[33];
  p.op_types = (const int*)d_in[34];
  p.labels   = (const int*)d_in[35];
  p.out      = (float*)d_out;

  zero_out_kernel<<<1, 64, 0, stream>>>(p.out);
  transformer_crf_kernel<<<1024, 256, 0, stream>>>(p);
}
